// LSTM_50508815401387
// MI455X (gfx1250) — compile-verified, matrix operand order checked by reading, not run on hardware
//
#include <hip/hip_runtime.h>

// Problem constants
#define BB 64          // batch
#define TT 784         // timesteps
#define HH 600         // hidden
#define HP 608         // hidden padded to 38*16 (K padded to 19*32)
#define SP 616         // LDS row stride in shorts (padded, 16B-aligned rows)
#define KT 19          // K tiles of 32
#define NWG 38         // one WG per 16-hidden-unit block
#define NTHREADS 512   // 16 wave32s: 4 M-tiles x 4 gate-tiles

typedef __attribute__((ext_vector_type(16))) __bf16 v16bf;
typedef __attribute__((ext_vector_type(8)))  float  v8f;

union frag { uint4 u[2]; v16bf v; };

__device__ __forceinline__ unsigned short f2bf(float f) {
    unsigned u = __builtin_bit_cast(unsigned, f);
    u += 0x7fffu + ((u >> 16) & 1u);          // round-to-nearest-even
    return (unsigned short)(u >> 16);
}
__device__ __forceinline__ float sigf(float v) { return 1.0f / (1.0f + __expf(-v)); }

// Build bf16 W^T (k-contiguous per gate column, zero-padded), bf16 initial h,
// and time-major xT[t][b].
__global__ void lstm_init(const float* __restrict__ hs0,
                          const float* __restrict__ W_hh,
                          const float* __restrict__ x,
                          unsigned short* __restrict__ hbuf0,
                          unsigned short* __restrict__ wt,
                          float* __restrict__ xT) {
    long i = (long)blockIdx.x * blockDim.x + threadIdx.x;
    const long nwt = (long)4 * HP * HP;
    const long nh  = (long)BB * HP;
    if (i < nwt) {
        int col = (int)(i / HP), k = (int)(i % HP);
        int gt = col / HP, j = col % HP;
        float v = (j < HH && k < HH) ? W_hh[((size_t)(gt * HH + j)) * HH + k] : 0.0f;
        wt[i] = f2bf(v);
    } else if (i < nwt + nh) {
        long r = i - nwt;
        int b = (int)(r / HP), k = (int)(r % HP);
        hbuf0[r] = (k < HH) ? f2bf(hs0[b * HH + k]) : (unsigned short)0;
    } else if (i < nwt + nh + (long)TT * BB) {
        long r = i - nwt - nh;
        int t = (int)(r / BB), b = (int)(r % BB);
        xT[r] = x[(size_t)b * TT + t];
    }
}

// Persistent cooperative LSTM scan. Weights register-resident; h staged in LDS
// via CDNA5 async global->LDS copies; A loads pipelined over two WMMA chains.
__global__ __launch_bounds__(NTHREADS, 1) void lstm_scan(
        const float* __restrict__ xT,      // [T][B]
        const float* __restrict__ cs0,     // [B,H]
        const float* __restrict__ W_ih,    // [4H,1]
        const float* __restrict__ b_ih,    // [4H]
        const float* __restrict__ b_hh,    // [4H]
        const unsigned short* __restrict__ wt,   // [4*HP][HP] bf16 W^T
        unsigned short* __restrict__ hbuf,       // [2][B][HP] bf16 double buffer
        float* __restrict__ hfinal,              // [B][H]
        unsigned int* __restrict__ ctr) {

    __shared__ unsigned short hLDS[BB * SP]; // staged A matrix (h), padded rows
    __shared__ float gatesLDS[BB * 4 * 16];  // [b][gate][col]
    __shared__ float cLDS[BB * 16];          // [b][col] cell state slice
    __shared__ float biasLDS[4 * 16];
    __shared__ float wihLDS[4 * 16];

    const int tid   = threadIdx.x;
    const int wg    = blockIdx.x;           // hidden block: j = wg*16 + col
    const int lane  = tid & 31;
    const int wid   = tid >> 5;             // 0..15
    const int mt    = wid >> 2;             // M tile 0..3 (batch rows)
    const int gt    = wid & 3;              // gate 0..3 (i,f,g,o)
    const int lrow  = lane & 15;
    const int lhalf = lane >> 4;

    if (tid < 64) {
        int g = tid >> 4, c = tid & 15;
        int j = wg * 16 + c;
        float bs = 0.0f, wi = 0.0f;
        if (j < HH) { bs = b_ih[g * HH + j] + b_hh[g * HH + j]; wi = W_ih[g * HH + j]; }
        biasLDS[tid] = bs; wihLDS[tid] = wi;
    }
    for (int e = tid; e < BB * 16; e += NTHREADS) {
        int b = e >> 4, c = e & 15;
        int j = wg * 16 + c;
        cLDS[e] = (j < HH) ? cs0[b * HH + j] : 0.0f;
    }

    // ---- B fragments: load once, keep in VGPRs for the entire scan ----
    frag Bf[KT];
    {
        const unsigned short* wtcol =
            wt + ((size_t)(gt * HP + wg * 16 + lrow)) * HP + (size_t)lhalf * 8;
        #pragma unroll
        for (int kt = 0; kt < KT; ++kt) {
            Bf[kt].u[0] = *(const uint4*)(wtcol + kt * 32);
            Bf[kt].u[1] = *(const uint4*)(wtcol + kt * 32 + 16);
        }
    }
    __syncthreads();

    for (int t = 0; t < TT; ++t) {
        const unsigned short* hcur = hbuf + (size_t)(t & 1) * BB * HP;
        unsigned short*       hnxt = hbuf + (size_t)((t + 1) & 1) * BB * HP;

        // ---- stage h into LDS: async global->LDS b128 (ASYNCcnt tracked) ----
        for (int i = tid; i < BB * (HP / 8); i += NTHREADS) {
            int row = i / (HP / 8), ch = i - row * (HP / 8);
            unsigned ldsOff = (unsigned)(unsigned long long)
                              (const void*)&hLDS[row * SP + ch * 8];
            unsigned long long ga = (unsigned long long)
                                    (const void*)(hcur + (size_t)i * 8);
            asm volatile("global_load_async_to_lds_b128 %0, %1, off"
                         :: "v"(ldsOff), "v"(ga) : "memory");
        }
        asm volatile("s_wait_asynccnt 0x0" ::: "memory");
        __syncthreads();

        // ---- GEMM phase: two independent WMMA chains (even/odd k-tiles),
        //      4 A buffers, prefetch distance = 2 pairs (4 WMMAs) ----
        const unsigned short* aLDS = &hLDS[(mt * 16 + lrow) * SP + lhalf * 8];
        frag fa[2][2];                       // [chain][phase]
        #define LDA(dst, kt)  do { \
            (dst).u[0] = *(const uint4*)(aLDS + (kt) * 32); \
            (dst).u[1] = *(const uint4*)(aLDS + (kt) * 32 + 16); } while (0)
        LDA(fa[0][0], 0); LDA(fa[1][0], 1);
        LDA(fa[0][1], 2); LDA(fa[1][1], 3);
        v8f acc0 = {}, acc1 = {};
        #pragma unroll
        for (int p = 0; p < 10; ++p) {       // pair p -> k-tiles 2p, 2p+1
            const int ph = p & 1;
            acc0 = __builtin_amdgcn_wmma_f32_16x16x32_bf16(
                       false, fa[0][ph].v, false, Bf[2 * p].v,
                       (short)0, acc0, false, false);
            if (2 * p + 1 < KT)
                acc1 = __builtin_amdgcn_wmma_f32_16x16x32_bf16(
                           false, fa[1][ph].v, false, Bf[2 * p + 1].v,
                           (short)0, acc1, false, false);
            if (2 * p + 4 < KT) LDA(fa[0][ph], 2 * p + 4);
            if (2 * p + 5 < KT) LDA(fa[1][ph], 2 * p + 5);
        }
        #undef LDA
        v8f acc = acc0 + acc1;

        // C/D layout: VGPR v -> row mt*16 + lhalf*8 + v, col lrow
        #pragma unroll
        for (int v = 0; v < 8; ++v) {
            int b = mt * 16 + lhalf * 8 + v;
            gatesLDS[(b * 4 + gt) * 16 + lrow] = acc[v];
        }
        __syncthreads();

        // ---- cell phase: 1024 (b,col) elems; 2 adjacent cols per thread ----
        {
            int e2 = tid * 2;
            int b  = e2 >> 4;
            int c  = e2 & 15;                  // even
            float xv = xT[t * BB + b];
            const float* gb = &gatesLDS[b * 64];
            float h2[2];
            #pragma unroll
            for (int r = 0; r < 2; ++r) {
                int cc0 = c + r;
                float gi = gb[cc0]      + xv * wihLDS[cc0]      + biasLDS[cc0];
                float gf = gb[16 + cc0] + xv * wihLDS[16 + cc0] + biasLDS[16 + cc0];
                float gg = gb[32 + cc0] + xv * wihLDS[32 + cc0] + biasLDS[32 + cc0];
                float go = gb[48 + cc0] + xv * wihLDS[48 + cc0] + biasLDS[48 + cc0];
                float cv = sigf(gf) * cLDS[e2 + r] + sigf(gi) * tanhf(gg);
                float hv = sigf(go) * tanhf(cv);
                cLDS[e2 + r] = cv;
                h2[r] = hv;
                if (t == TT - 1) {
                    int j = wg * 16 + cc0;
                    if (j < HH) hfinal[b * HH + j] = hv;
                }
            }
            unsigned packed = (unsigned)f2bf(h2[0]) | ((unsigned)f2bf(h2[1]) << 16);
            *(unsigned*)&hnxt[b * HP + wg * 16 + c] = packed;
        }
        __syncthreads();

        // ---- grid-wide barrier (monotonic counter, reset each launch) ----
        __threadfence();
        if (tid == 0) {
            __hip_atomic_fetch_add(ctr, 1u, __ATOMIC_ACQ_REL, __HIP_MEMORY_SCOPE_AGENT);
            unsigned target = (unsigned)(t + 1) * (unsigned)gridDim.x;
            while (__hip_atomic_load(ctr, __ATOMIC_ACQUIRE, __HIP_MEMORY_SCOPE_AGENT) < target) {
                __builtin_amdgcn_s_sleep(2);
            }
        }
        __syncthreads();
        __threadfence();
    }
}

// out[b,o] = hfinal[b,:] . W_out[o,:] + b_out[o]   (64x10, trivial)
__global__ void lstm_out(const float* __restrict__ hfinal,
                         const float* __restrict__ W_out,
                         const float* __restrict__ b_out,
                         float* __restrict__ out) {
    int idx = blockIdx.x * blockDim.x + threadIdx.x;
    if (idx >= BB * 10) return;
    int b = idx / 10, o = idx % 10;
    float s = b_out[o];
    const float* hp = hfinal + (size_t)b * HH;
    const float* wp = W_out + (size_t)o * HH;
    for (int k = 0; k < HH; ++k) s = fmaf(hp[k], wp[k], s);
    out[idx] = s;
}

extern "C" void kernel_launch(void* const* d_in, const int* in_sizes, int n_in,
                              void* d_out, int out_size, void* d_ws, size_t ws_size,
                              hipStream_t stream) {
    const float* x     = (const float*)d_in[0];
    const float* hs0   = (const float*)d_in[1];
    const float* cs0   = (const float*)d_in[2];
    const float* W_ih  = (const float*)d_in[3];
    const float* W_hh  = (const float*)d_in[4];
    const float* b_ih  = (const float*)d_in[5];
    const float* b_hh  = (const float*)d_in[6];
    const float* W_out = (const float*)d_in[7];
    const float* b_out = (const float*)d_in[8];
    float* out = (float*)d_out;

    char* ws = (char*)d_ws;
    size_t off = 0;
    unsigned short* wt = (unsigned short*)(ws + off);
    off += (size_t)4 * HP * HP * 2;                  // 2,957,312 B
    unsigned short* hbuf = (unsigned short*)(ws + off);
    size_t hbytes = (size_t)2 * BB * HP * 2;         // 155,648 B
    off += hbytes;
    float* hfinal = (float*)(ws + off);
    size_t fbytes = (size_t)BB * HH * 4;             // 153,600 B
    off += fbytes;
    unsigned int* ctr = (unsigned int*)(ws + off);
    off += 256;
    float* xT = (float*)(ws + off);
    off += (size_t)TT * BB * 4;                      // 200,704 B

    // Reset h double-buffer (incl. pad columns), hfinal, and barrier counter
    // every launch: deterministic and graph-capture safe (memset node).
    hipMemsetAsync(hbuf, 0, hbytes + fbytes + 256, stream);

    long ninit = (long)4 * HP * HP + (long)BB * HP + (long)TT * BB;
    int ib = (int)((ninit + 255) / 256);
    lstm_init<<<ib, 256, 0, stream>>>(hs0, W_hh, x, hbuf, wt, xT);
    lstm_scan<<<NWG, NTHREADS, 0, stream>>>(xT, cs0, W_ih, b_ih, b_hh,
                                            wt, hbuf, hfinal, ctr);
    lstm_out<<<(BB * 10 + 255) / 256, 256, 0, stream>>>(hfinal, W_out, b_out, out);
}
